// CurricularFace_11330123727281
// MI455X (gfx1250) — matrix operand order repaired and checked by reading, compiled, not run on hardware
//
#include <hip/hip_runtime.h>
#include <hip/hip_bf16.h>

// ---------------- problem constants ----------------
#define EMB_N   1024          // rows (batch)
#define EMB_K   512           // feature dim
#define NCLS    100000        // classes
#define NPAD    100096        // NCLS padded to multiple of 128 (block N tile)
#define SCALE_S 64.0f
#define COS_M_C   0.8775825618903728f   // cos(0.5)
#define SIN_M_C   0.4794255386042030f   // sin(0.5)
#define THRESH_C (-0.8775825618903728f) // cos(pi-0.5)
#define MM_C      0.2397127693021015f   // sin(pi-0.5)*0.5

typedef __attribute__((ext_vector_type(16))) __bf16 v16bf;
typedef __attribute__((ext_vector_type(8)))  __bf16 v8bf;
typedef __attribute__((ext_vector_type(4)))  __bf16 v4bf;
typedef __attribute__((ext_vector_type(8)))  float  v8f;

// ---------------- workspace layout (bytes) ----------------
// [0, 1 MiB)                 : embn  bf16 [1024][512]
// [1 MiB, +102,498,304)      : knb   bf16 [512][NPAD]  (row-major along N)
// then: target_logit f32[1024], ctm f32[1024], fin f32[1024], tnew f32[1]
#define OFF_EMBN 0
#define OFF_KNB  (1048576)
#define OFF_TL   (OFF_KNB + (size_t)EMB_K * NPAD * 2)
#define OFF_CTM  (OFF_TL + 4096)
#define OFF_FIN  (OFF_CTM + 4096)
#define OFF_TNEW (OFF_FIN + 4096)

// ============ K1: L2-normalize embedding rows -> bf16 ============
__global__ __launch_bounds__(128) void k_norm_emb(const float* __restrict__ emb,
                                                  __bf16* __restrict__ embn) {
  const int row = blockIdx.x;           // 0..1023
  const int tid = threadIdx.x;          // 0..127, 4 floats each
  const float4 v = ((const float4*)(emb + (size_t)row * EMB_K))[tid];
  float ss = v.x * v.x + v.y * v.y + v.z * v.z + v.w * v.w;

  __shared__ float red[128];
  red[tid] = ss;
  __syncthreads();
  for (int s = 64; s > 0; s >>= 1) {
    if (tid < s) red[tid] += red[tid + s];
    __syncthreads();
  }
  __shared__ float sinv;
  if (tid == 0) sinv = rsqrtf(fmaxf(red[0], 1e-24f));
  __syncthreads();
  const float inv = sinv;

  v4bf o;
  o[0] = (__bf16)(v.x * inv);
  o[1] = (__bf16)(v.y * inv);
  o[2] = (__bf16)(v.z * inv);
  o[3] = (__bf16)(v.w * inv);
  *(v4bf*)(embn + (size_t)row * EMB_K + tid * 4) = o;
}

// ===== K2: L2-normalize kernel columns -> bf16 [512][NPAD], zero pad =====
__global__ __launch_bounds__(256) void k_norm_ker(const float* __restrict__ ker,
                                                  __bf16* __restrict__ knb) {
  const int c = blockIdx.x * 256 + threadIdx.x;   // column; grid covers NPAD exactly
  if (c >= NCLS) {                                // zero-fill pad columns
    for (int k = 0; k < EMB_K; ++k) knb[(size_t)k * NPAD + c] = (__bf16)0.0f;
    return;
  }
  float ss = 0.0f;
  for (int k = 0; k < EMB_K; ++k) {
    const float v = ker[(size_t)k * NCLS + c];    // coalesced across threads
    ss += v * v;
  }
  const float inv = rsqrtf(fmaxf(ss, 1e-24f));
  for (int k = 0; k < EMB_K; ++k) {
    knb[(size_t)k * NPAD + c] = (__bf16)(ker[(size_t)k * NCLS + c] * inv);
  }
}

// ============ K3: target logit per row (deterministic) ============
__global__ __launch_bounds__(128) void k_target(const __bf16* __restrict__ embn,
                                                const __bf16* __restrict__ knb,
                                                const int* __restrict__ label,
                                                float* __restrict__ target_logit) {
  const int row = blockIdx.x;
  const int tid = threadIdx.x;
  const int lab = label[row];
  float s = 0.0f;
  for (int k = tid; k < EMB_K; k += 128)
    s += (float)embn[(size_t)row * EMB_K + k] * (float)knb[(size_t)k * NPAD + lab];

  __shared__ float red[128];
  red[tid] = s;
  __syncthreads();
  for (int st = 64; st > 0; st >>= 1) {
    if (tid < st) red[tid] += red[tid + st];
    __syncthreads();
  }
  if (tid == 0) target_logit[row] = fminf(1.0f, fmaxf(-1.0f, red[0]));
}

// ==== K4: t_new (deterministic tree reduce) + per-row margin terms ====
__global__ __launch_bounds__(1024) void k_stats(const float* __restrict__ target_logit,
                                                const float* __restrict__ t_in,
                                                float* __restrict__ ctm,
                                                float* __restrict__ fin,
                                                float* __restrict__ tnew_out) {
  const int tid = threadIdx.x;                    // 0..1023 == row
  const float tl = target_logit[tid];

  __shared__ float red[1024];
  red[tid] = tl;
  __syncthreads();
  for (int s = 512; s > 0; s >>= 1) {
    if (tid < s) red[tid] += red[tid + s];
    __syncthreads();
  }
  if (tid == 0)
    tnew_out[0] = red[0] * (1.0f / 1024.0f) * 0.01f + 0.99f * t_in[0];

  const float sn = sqrtf(fmaxf(0.0f, 1.0f - tl * tl));
  const float cm = tl * COS_M_C - sn * SIN_M_C;
  ctm[tid] = cm;
  fin[tid] = (tl > THRESH_C) ? cm : (tl - MM_C);
}

// ============ K5: bf16 WMMA GEMM + CurricularFace epilogue ============
// Block: 256 threads = 8 waves, arranged 4(M) x 2(N). Block tile 64M x 128N.
// Wave tile 16M x 64N = 4 accumulators of v_wmma_f32_16x16x32_bf16.
__global__ __launch_bounds__(256) void k_gemm(const __bf16* __restrict__ embn,
                                              const __bf16* __restrict__ knb,
                                              const float* __restrict__ ctm,
                                              const float* __restrict__ fin,
                                              const float* __restrict__ tnew_p,
                                              const int* __restrict__ label,
                                              float* __restrict__ out) {
  const int lane = threadIdx.x & 31;
  const int wave = threadIdx.x >> 5;
  const int wm = wave & 3;        // 0..3  -> M sub-tile
  const int wn = wave >> 2;       // 0..1  -> N sub-tile
  const int m_base = blockIdx.y * 64 + wm * 16;
  const int n_base = blockIdx.x * 128 + wn * 64;
  const int lh = lane >> 4;       // lane half (A: K phase / B: K+8)
  const int ll = lane & 15;

  v8f acc0 = {}, acc1 = {}, acc2 = {}, acc3 = {};

  // A fragment (16x32 bf16 layout): lane l holds row M=l%16;
  //   low lanes: K = kk+0..7 and kk+16..23; high lanes: K = kk+8..15 and kk+24..31.
  const __bf16* aptr = embn + (size_t)(m_base + ll) * EMB_K + 8 * lh;
  // B fragment (32x16 bf16 layout): lane l holds column N=l%16;
  //   low lanes: K = kk+0..7 and kk+16..23; high lanes: K = kk+8..15 and kk+24..31.
  const __bf16* bptr = knb + (size_t)(8 * lh) * NPAD + n_base + ll;

  for (int kk = 0; kk < EMB_K; kk += 32) {
    const v8bf alo = *(const v8bf*)(aptr);
    const v8bf ahi = *(const v8bf*)(aptr + 16);
    v16bf a;
#pragma unroll
    for (int i = 0; i < 8; ++i) { a[i] = alo[i]; a[i + 8] = ahi[i]; }

    v16bf b0, b1, b2, b3;
#pragma unroll
    for (int i = 0; i < 8; ++i) {
      const __bf16* p0 = bptr + (size_t)i * NPAD;
      const __bf16* p1 = bptr + (size_t)(16 + i) * NPAD;
      b0[i] = p0[0];  b0[i + 8] = p1[0];
      b1[i] = p0[16]; b1[i + 8] = p1[16];
      b2[i] = p0[32]; b2[i + 8] = p1[32];
      b3[i] = p0[48]; b3[i + 8] = p1[48];
    }

    if (kk < EMB_K - 32) {   // pull next K-slab of B toward L2/L0 (global_prefetch_b8)
      __builtin_prefetch((const void*)(bptr + (size_t)32 * NPAD), 0, 1);
    }

    acc0 = __builtin_amdgcn_wmma_f32_16x16x32_bf16(false, a, false, b0, (short)0, acc0, false, false);
    acc1 = __builtin_amdgcn_wmma_f32_16x16x32_bf16(false, a, false, b1, (short)0, acc1, false, false);
    acc2 = __builtin_amdgcn_wmma_f32_16x16x32_bf16(false, a, false, b2, (short)0, acc2, false, false);
    acc3 = __builtin_amdgcn_wmma_f32_16x16x32_bf16(false, a, false, b3, (short)0, acc3, false, false);

    aptr += 32;
    bptr += (size_t)32 * NPAD;
  }

  // Epilogue. C/D layout: VGPR r -> M = m_base + r + 8*lh, N = n_base + 16*t + ll.
  const float tnew = tnew_p[0];
  const int m0 = m_base + 8 * lh;
  float cm_r[8], fi_r[8];
  int lb_r[8];
#pragma unroll
  for (int r = 0; r < 8; ++r) {
    cm_r[r] = ctm[m0 + r];
    fi_r[r] = fin[m0 + r];
    lb_r[r] = label[m0 + r];
  }

  float accs[4][8];
#pragma unroll
  for (int r = 0; r < 8; ++r) {
    accs[0][r] = acc0[r]; accs[1][r] = acc1[r];
    accs[2][r] = acc2[r]; accs[3][r] = acc3[r];
  }

#pragma unroll
  for (int t = 0; t < 4; ++t) {
    const int n = n_base + 16 * t + ll;
    if (n < NCLS) {
#pragma unroll
      for (int r = 0; r < 8; ++r) {
        float c = fminf(1.0f, fmaxf(-1.0f, accs[t][r]));
        if (c > cm_r[r]) c = c * (tnew + c);          // hard-example curriculum
        if (n == lb_r[r]) c = fi_r[r];                // target-column override
        out[(size_t)(m0 + r) * NCLS + n] = c * SCALE_S;
      }
    }
  }
}

// ---------------- host launcher ----------------
extern "C" void kernel_launch(void* const* d_in, const int* in_sizes, int n_in,
                              void* d_out, int out_size, void* d_ws, size_t ws_size,
                              hipStream_t stream) {
  const float* emb   = (const float*)d_in[0];   // [1024,512]
  const float* ker   = (const float*)d_in[1];   // [512,100000]
  const float* t_in  = (const float*)d_in[2];   // [1]
  const int*   label = (const int*)d_in[3];     // [1024]
  float* out = (float*)d_out;                   // [1024,100000]

  char* ws = (char*)d_ws;
  __bf16* embn = (__bf16*)(ws + OFF_EMBN);
  __bf16* knb  = (__bf16*)(ws + OFF_KNB);
  float*  tl   = (float*)(ws + OFF_TL);
  float*  ctm  = (float*)(ws + OFF_CTM);
  float*  fin  = (float*)(ws + OFF_FIN);
  float*  tnew = (float*)(ws + OFF_TNEW);

  k_norm_emb<<<EMB_N, 128, 0, stream>>>(emb, embn);
  k_norm_ker<<<NPAD / 256, 256, 0, stream>>>(ker, knb);
  k_target<<<EMB_N, 128, 0, stream>>>(embn, knb, label, tl);
  k_stats<<<1, 1024, 0, stream>>>(tl, t_in, ctm, fin, tnew);

  dim3 grid(NPAD / 128, EMB_N / 64, 1);         // 782 x 16 blocks
  k_gemm<<<grid, 256, 0, stream>>>(embn, knb, ctm, fin, tnew, label, out);
}
